// MoEAttention_46901042872834
// MI455X (gfx1250) — matrix operand order repaired and checked by reading, not compiled, not run on hardware
//
#include <hip/hip_runtime.h>
#include <hip/hip_bf16.h>

// ---------------- problem constants ----------------
#define B_  2
#define S_  2048
#define D_  1024
#define E_  4
#define H_  16
#define HD_ 64
#define M_  (B_*S_)   // 4096 tokens

typedef __attribute__((ext_vector_type(16))) __bf16 bf16x16;
typedef __attribute__((ext_vector_type(8)))  __bf16 bf16x8;
typedef __attribute__((ext_vector_type(4)))  __bf16 bf16x4;
typedef __attribute__((ext_vector_type(8)))  float  f32x8;

#define WMMA_BF16(a,b,c) \
  __builtin_amdgcn_wmma_f32_16x16x32_bf16(false,(a),false,(b),(short)0,(c),false,false)

// gfx1250 async global->LDS copies (ASYNCcnt), guarded for toolchain support
#if defined(__has_builtin)
#if __has_builtin(__builtin_amdgcn_global_load_async_to_lds_b128) && \
    __has_builtin(__builtin_amdgcn_s_wait_asynccnt)
#define USE_ASYNC_LDS 1
#endif
#endif
#ifndef USE_ASYNC_LDS
#define USE_ASYNC_LDS 0
#endif

#if USE_ASYNC_LDS
// Builtin signature (from hipcc diagnostic): both pointer params are
// non-const pointers to `int __attribute__((vector_size(16)))` — global (AS1)
// source, LDS (AS3) destination — plus imm offset and imm cpol.
typedef int v4i_ __attribute__((__vector_size__(16)));
#define ASYNC_B128(g, l)                                                       \
  __builtin_amdgcn_global_load_async_to_lds_b128(                              \
      (__attribute__((address_space(1))) v4i_*)(void*)(const void*)(g),        \
      (__attribute__((address_space(3))) v4i_*)(void*)(l), 0, 0)
#endif

// A-fragment (16x32 bf16): lane = row (lane&15); two contiguous 8-elem chunks:
//   K = 8*(lane>>4)+i (i<8) and K = 16+8*(lane>>4)+(i-8) (i>=8)
__device__ __forceinline__ bf16x16 ldA(const __bf16* p) {
  bf16x8 lo = *(const bf16x8*)(p);
  bf16x8 hi = *(const bf16x8*)(p + 16);
  bf16x16 r;
#pragma unroll
  for (int i = 0; i < 8; ++i) { r[i] = lo[i]; r[i + 8] = hi[i]; }
  return r;
}

// ---------------- generic bf16 GEMM core ----------------
// C(128x128 per block) = A(MxK row-major) * Bm(NxK row-major)^T
// 8 waves: wave tile 32 rows x 64 cols (2x4 accumulators).
__device__ __forceinline__ void gemm_core(const __bf16* __restrict__ A,
                                          const __bf16* __restrict__ Bm,
                                          int K, int row0, int col0, int lane,
                                          f32x8 acc[2][4]) {
  const int mloc = lane & 15;
  const int ka = (lane >> 4) << 3;   // A half-K offset
  const int kb = (lane >> 4) << 3;   // B half-K offset (same split as A)
  const __bf16* a0p = A  + (size_t)(row0 + mloc) * K + ka;
  const __bf16* a1p = a0p + (size_t)16 * K;
  const __bf16* b0p = Bm + (size_t)(col0 + mloc) * K + kb;
  for (int k = 0; k < K; k += 32) {
    __builtin_prefetch(a0p + k + 256, 0, 1);   // global_prefetch_b8
    bf16x16 a0 = ldA(a0p + k);
    bf16x16 a1 = ldA(a1p + k);
#pragma unroll
    for (int j = 0; j < 4; ++j) {
      bf16x16 b = ldA(b0p + (size_t)(j * 16) * K + k);
      acc[0][j] = WMMA_BF16(a0, b, acc[0][j]);
      acc[1][j] = WMMA_BF16(a1, b, acc[1][j]);
    }
  }
}

// ---------------- GEMM: fp32 row-major output (Q/K projections) ----------------
__global__ __launch_bounds__(256) void gemm_f32_kernel(const __bf16* __restrict__ A,
                                                       const __bf16* __restrict__ Bm,
                                                       float* __restrict__ C,
                                                       int K, int N) {
  int lane = threadIdx.x & 31, wid = threadIdx.x >> 5;
  int row0 = blockIdx.y * 128 + (wid >> 1) * 32;
  int col0 = blockIdx.x * 128 + (wid & 1) * 64;
  f32x8 acc[2][4] = {};
  gemm_core(A, Bm, K, row0, col0, lane, acc);
  int n0 = lane & 15, r0 = (lane >> 4) << 3;
#pragma unroll
  for (int i = 0; i < 2; ++i)
#pragma unroll
    for (int j = 0; j < 4; ++j) {
      int col = col0 + j * 16 + n0;
      int row = row0 + i * 16 + r0;
#pragma unroll
      for (int r = 0; r < 8; ++r)
        C[(size_t)(row + r) * N + col] = acc[i][j][r];
    }
}

// ---------------- GEMM: V projection, writes V^T as bf16 vt[b][h][d][s] ----------
__global__ __launch_bounds__(256) void gemm_vt_kernel(const __bf16* __restrict__ A,
                                                      const __bf16* __restrict__ Bm,
                                                      __bf16* __restrict__ vt, int K) {
  int lane = threadIdx.x & 31, wid = threadIdx.x >> 5;
  int row0 = blockIdx.y * 128 + (wid >> 1) * 32;
  int col0 = blockIdx.x * 128 + (wid & 1) * 64;
  f32x8 acc[2][4] = {};
  gemm_core(A, Bm, K, row0, col0, lane, acc);
  int n0 = lane & 15, r0 = (lane >> 4) << 3;
#pragma unroll
  for (int i = 0; i < 2; ++i)
#pragma unroll
    for (int j = 0; j < 4; ++j) {
      int n  = col0 + j * 16 + n0;        // output dim 0..1023
      int m0 = row0 + i * 16 + r0;        // 8 consecutive tokens
      int bb = m0 >> 11, s0 = m0 & (S_ - 1);
      __bf16* dst = vt + ((size_t)(bb * H_ + (n >> 6)) * HD_ + (n & 63)) * S_ + s0;
      bf16x8 pk;
#pragma unroll
      for (int r = 0; r < 8; ++r) pk[r] = (__bf16)acc[i][j][r];
      *(bf16x8*)dst = pk;                 // 16B contiguous along s
    }
}

// ---------------- GEMM: output projection, gated accumulate into d_out ----------
__global__ __launch_bounds__(256) void gemm_gated_kernel(const __bf16* __restrict__ A,
                                                         const __bf16* __restrict__ Bm,
                                                         const float* __restrict__ gate,
                                                         float* __restrict__ out,
                                                         int e, int K, int N) {
  int lane = threadIdx.x & 31, wid = threadIdx.x >> 5;
  int row0 = blockIdx.y * 128 + (wid >> 1) * 32;
  int col0 = blockIdx.x * 128 + (wid & 1) * 64;
  f32x8 acc[2][4] = {};
  gemm_core(A, Bm, K, row0, col0, lane, acc);
  int n0 = lane & 15, r0 = (lane >> 4) << 3;
#pragma unroll
  for (int i = 0; i < 2; ++i)
#pragma unroll
    for (int j = 0; j < 4; ++j) {
      int col = col0 + j * 16 + n0;
      int row = row0 + i * 16 + r0;
#pragma unroll
      for (int r = 0; r < 8; ++r) {
        float g = gate[(size_t)(row + r) * E_ + e];
        size_t o = (size_t)(row + r) * N + col;
        if (e == 0) out[o] = g * acc[i][j][r];
        else        out[o] += g * acc[i][j][r];
      }
    }
}

// ---------------- WMMA flash attention ----------------
// S^T = K*Q^T so the score accumulator layout IS the A-fragment layout for P*V
// (no LDS transpose). 4 waves/block share one (b,h): K/V k-tiles are staged in
// LDS with double-buffered global_load_async_to_lds_b128 (ASYNCcnt) so waves
// don't redundantly load the same tiles and the async fill overlaps compute.
__global__ __launch_bounds__(128) void flash_kernel(const __bf16* __restrict__ qh,
                                                    const __bf16* __restrict__ kh,
                                                    const __bf16* __restrict__ vt,
                                                    __bf16* __restrict__ oh) {
  int lane = threadIdx.x & 31;
  int wv = threadIdx.x >> 5;             // 0..3
  int b = blockIdx.z, h = blockIdx.y;
  int q0 = blockIdx.x * 64 + wv * 16;

  // Q as B-fragment of S^T: lane col = q token, d in A-fragment split order
  int qt = q0 + (lane & 15);
  const __bf16* qp = qh + ((size_t)b * S_ + qt) * D_ + h * HD_ + ((lane >> 4) << 3);
  bf16x16 qb0 = ldA(qp);          // d 0..31
  bf16x16 qb1 = ldA(qp + 32);   // d 32..63

  const __bf16* kbase = kh + (size_t)b * S_ * D_ + h * HD_;
  const __bf16* vbase = vt + (size_t)(b * H_ + h) * HD_ * S_;
  int rowoff = lane & 15;
  int ka8 = (lane >> 4) << 3;
  int vk16 = (lane >> 4) << 3;
  int srcb = (lane >> 4) << 3;

  f32x8 o0 = {}, o1 = {}, o2 = {}, o3 = {};
  f32x8 zero = {};
  float m_run = -1e30f, l_run = 0.f;

#if USE_ASYNC_LDS
  __shared__ __align__(32) __bf16 kt[2][32][64];   // [buf][k-row][d]
  __shared__ __align__(32) __bf16 vs[2][64][32];   // [buf][d][k]
  int tid = threadIdx.x;
  // Each wave issues exactly 4 async b128 per tile -> s_wait_asynccnt(4)
  // guarantees the previous tile landed while the next stays in flight.
  auto issue_tile = [&](int buf, int kk) {
#pragma unroll
    for (int u = 0; u < 2; ++u) {                  // K tile: 32x64 bf16 = 4KB
      int c = tid + u * 128;
      int row = c >> 3, d8 = (c & 7) << 3;
      ASYNC_B128(kbase + (size_t)(kk + row) * D_ + d8, &kt[buf][row][d8]);
    }
#pragma unroll
    for (int u = 0; u < 2; ++u) {                  // V tile: 64x32 bf16 = 4KB
      int c = tid + u * 128;
      int dd = c >> 2, k8 = (c & 3) << 3;
      ASYNC_B128(vbase + (size_t)dd * S_ + kk + k8, &vs[buf][dd][k8]);
    }
  };
  issue_tile(0, 0);
  int buf = 0;
#endif

  for (int kk = 0; kk < S_; kk += 32) {
#if USE_ASYNC_LDS
    if (kk + 32 < S_) {
      issue_tile(buf ^ 1, kk + 32);
      __builtin_amdgcn_s_wait_asynccnt(4);
    } else {
      __builtin_amdgcn_s_wait_asynccnt(0);
    }
    __syncthreads();                                // all waves' deposits visible
    const __bf16* kr0 = &kt[buf][rowoff][ka8];
    const __bf16* kr1 = &kt[buf][16 + rowoff][ka8];
    bf16x16 a00 = ldA(kr0);                         // ds_load_b128 x2
    bf16x16 a01 = ldA(kr0 + 32);
    bf16x16 a10 = ldA(kr1);
    bf16x16 a11 = ldA(kr1 + 32);
#else
    const __bf16* kp0 = kbase + (size_t)(kk + rowoff) * D_ + ka8;
    const __bf16* kp1 = kp0 + (size_t)16 * D_;
    bf16x16 a00 = ldA(kp0);
    bf16x16 a01 = ldA(kp0 + 32);
    bf16x16 a10 = ldA(kp1);
    bf16x16 a11 = ldA(kp1 + 32);
#endif

    f32x8 st0 = WMMA_BF16(a00, qb0, zero); st0 = WMMA_BF16(a01, qb1, st0);
    f32x8 st1 = WMMA_BF16(a10, qb0, zero); st1 = WMMA_BF16(a11, qb1, st1);

    // online softmax stats; each lane owns one q column, 16 k values
    float mloc = fmaxf(st0[0], st1[0]);
#pragma unroll
    for (int r = 1; r < 8; ++r) mloc = fmaxf(mloc, fmaxf(st0[r], st1[r]));
    mloc = fmaxf(mloc, __shfl_xor(mloc, 16, 32));
    float m_new = fmaxf(m_run, mloc);

    float lloc = 0.f;
    bf16x16 pa;                    // directly in P*V A-fragment layout
#pragma unroll
    for (int r = 0; r < 8; ++r) {
      float e0 = __expf(st0[r] - m_new);
      float e1 = __expf(st1[r] - m_new);
      lloc += e0 + e1;
      pa[r]     = (__bf16)e0;
      pa[r + 8] = (__bf16)e1;
    }
    lloc += __shfl_xor(lloc, 16, 32);
    float alpha = __expf(m_run - m_new);
    l_run = l_run * alpha + lloc;
    m_run = m_new;

    // rescale O accumulators (per-q alpha redistributed to O layout)
    float a8[8];
#pragma unroll
    for (int r = 0; r < 8; ++r) a8[r] = __shfl(alpha, srcb + r, 32);
#pragma unroll
    for (int r = 0; r < 8; ++r) {
      o0[r] *= a8[r]; o1[r] *= a8[r]; o2[r] *= a8[r]; o3[r] *= a8[r];
    }

    // O += P * V
#if USE_ASYNC_LDS
    o0 = WMMA_BF16(pa, ldA(&vs[buf][(lane & 15)][vk16]),      o0);
    o1 = WMMA_BF16(pa, ldA(&vs[buf][16 + (lane & 15)][vk16]), o1);
    o2 = WMMA_BF16(pa, ldA(&vs[buf][32 + (lane & 15)][vk16]), o2);
    o3 = WMMA_BF16(pa, ldA(&vs[buf][48 + (lane & 15)][vk16]), o3);
    __syncthreads();               // done reading before next buffer overwrite
    buf ^= 1;
#else
    const __bf16* vp = vbase + (size_t)(lane & 15) * S_ + kk + vk16;
    o0 = WMMA_BF16(pa, ldA(vp),            o0);
    o1 = WMMA_BF16(pa, ldA(vp + 16 * S_),  o1);
    o2 = WMMA_BF16(pa, ldA(vp + 32 * S_),  o2);
    o3 = WMMA_BF16(pa, ldA(vp + 48 * S_),  o3);
#endif
  }

  float linv = 1.0f / l_run;
  float l8[8];
#pragma unroll
  for (int r = 0; r < 8; ++r) l8[r] = __shfl(linv, srcb + r, 32);

  int dcol = lane & 15;
  int mr0 = q0 + ((lane >> 4) << 3);
  __bf16* obase = oh + ((size_t)b * S_ + mr0) * D_ + h * HD_ + dcol;
#pragma unroll
  for (int r = 0; r < 8; ++r) {
    size_t ro = (size_t)r * D_;
    obase[ro]      = (__bf16)(o0[r] * l8[r]);
    obase[ro + 16] = (__bf16)(o1[r] * l8[r]);
    obase[ro + 32] = (__bf16)(o2[r] * l8[r]);
    obase[ro + 48] = (__bf16)(o3[r] * l8[r]);
  }
}

// ---------------- fused LayerNorm + RoPE (fp32 in -> bf16 out) ----------------
__global__ __launch_bounds__(256) void lnrope_kernel(const float* __restrict__ in,
                                                     const float* __restrict__ gamma,
                                                     const float* __restrict__ beta,
                                                     const float* __restrict__ fcos,
                                                     const float* __restrict__ fsin,
                                                     __bf16* __restrict__ out,
                                                     float scale) {
  __shared__ float ssum[8], ssq[8];
  int t = blockIdx.x;
  int s = t & (S_ - 1);
  int tid = threadIdx.x, lane = tid & 31, wid = tid >> 5;
  const float* row = in + (size_t)t * D_;
  float4 v = ((const float4*)row)[tid];
  float sum = v.x + v.y + v.z + v.w;
  float sq  = v.x * v.x + v.y * v.y + v.z * v.z + v.w * v.w;
#pragma unroll
  for (int off = 16; off; off >>= 1) {
    sum += __shfl_xor(sum, off, 32);
    sq  += __shfl_xor(sq,  off, 32);
  }
  if (lane == 0) { ssum[wid] = sum; ssq[wid] = sq; }
  __syncthreads();
  float ts = 0.f, tq = 0.f;
#pragma unroll
  for (int i = 0; i < 8; ++i) { ts += ssum[i]; tq += ssq[i]; }
  float mu = ts * (1.f / D_);
  float var = tq * (1.f / D_) - mu * mu;
  float rs = rsqrtf(var + 1e-5f);

  int d0 = tid * 4;
  float y0 = (v.x - mu) * rs * gamma[d0 + 0] + beta[d0 + 0];
  float y1 = (v.y - mu) * rs * gamma[d0 + 1] + beta[d0 + 1];
  float y2 = (v.z - mu) * rs * gamma[d0 + 2] + beta[d0 + 2];
  float y3 = (v.w - mu) * rs * gamma[d0 + 3] + beta[d0 + 3];

  int j0 = (d0 & 63) >> 1;
  float c0 = fcos[s * 32 + j0],     s0 = fsin[s * 32 + j0];
  float c1 = fcos[s * 32 + j0 + 1], s1 = fsin[s * 32 + j0 + 1];
  bf16x4 o4;
  o4[0] = (__bf16)((y0 * c0 - y1 * s0) * scale);
  o4[1] = (__bf16)((y0 * s0 + y1 * c0) * scale);
  o4[2] = (__bf16)((y2 * c1 - y3 * s1) * scale);
  o4[3] = (__bf16)((y2 * s1 + y3 * c1) * scale);
  ((bf16x4*)(out + (size_t)t * D_))[tid] = o4;
}

// ---------------- gate = sigmoid(x @ gate_w^T + gate_b) ----------------
__global__ __launch_bounds__(128) void gate_kernel(const float* __restrict__ x,
                                                   const float* __restrict__ gw,
                                                   const float* __restrict__ gb,
                                                   float* __restrict__ gate) {
  __shared__ float part[128];
  int t = blockIdx.x, tid = threadIdx.x;
  int e = tid & 3;
  const float* row = x + (size_t)t * D_;
  const float* w = gw + (size_t)e * D_;
  float acc = 0.f;
  for (int i = tid >> 2; i < D_; i += 32) acc += row[i] * w[i];
  part[tid] = acc;
  __syncthreads();
  if (tid < 4) {
    float ssum = 0.f;
#pragma unroll
    for (int k = 0; k < 32; ++k) ssum += part[tid + 4 * k];
    gate[(size_t)t * E_ + tid] = 1.f / (1.f + __expf(-(ssum + gb[tid])));
  }
}

// ---------------- fp32 -> bf16 conversion (vectorized) ----------------
__global__ __launch_bounds__(256) void cvt_kernel(const float* __restrict__ src,
                                                  __bf16* __restrict__ dst, int n4) {
  int i = blockIdx.x * blockDim.x + threadIdx.x;
  if (i < n4) {
    float4 v = ((const float4*)src)[i];
    bf16x4 o;
    o[0] = (__bf16)v.x; o[1] = (__bf16)v.y; o[2] = (__bf16)v.z; o[3] = (__bf16)v.w;
    ((bf16x4*)dst)[i] = o;
  }
}

// ---------------- host orchestration ----------------
extern "C" void kernel_launch(void* const* d_in, const int* in_sizes, int n_in,
                              void* d_out, int out_size, void* d_ws, size_t ws_size,
                              hipStream_t stream) {
  (void)in_sizes; (void)n_in; (void)out_size; (void)ws_size;
  const float* x       = (const float*)d_in[0];
  const float* fcos    = (const float*)d_in[1];
  const float* fsin    = (const float*)d_in[2];
  const float* wq      = (const float*)d_in[3];
  const float* wk      = (const float*)d_in[4];
  const float* wv      = (const float*)d_in[5];
  const float* wo      = (const float*)d_in[6];
  const float* q_gamma = (const float*)d_in[7];
  const float* q_beta  = (const float*)d_in[8];
  const float* k_gamma = (const float*)d_in[9];
  const float* k_beta  = (const float*)d_in[10];
  const float* gate_w  = (const float*)d_in[11];
  const float* gate_b  = (const float*)d_in[12];
  float* out = (float*)d_out;

  char* wsb = (char*)d_ws;
  size_t off = 0;
  auto alloc = [&](size_t bytes) { void* p = wsb + off; off += (bytes + 255) & ~(size_t)255; return p; };
  const size_t WB = (size_t)E_ * D_ * D_ * sizeof(__bf16);
  __bf16* whq = (__bf16*)alloc(WB);
  __bf16* whk = (__bf16*)alloc(WB);
  __bf16* whv = (__bf16*)alloc(WB);
  __bf16* who = (__bf16*)alloc(WB);
  __bf16* xh  = (__bf16*)alloc((size_t)M_ * D_ * sizeof(__bf16));
  float*  gtp = (float*) alloc((size_t)M_ * E_ * sizeof(float));
  float*  qf  = (float*) alloc((size_t)M_ * D_ * sizeof(float));
  float*  kf  = (float*) alloc((size_t)M_ * D_ * sizeof(float));
  __bf16* qh  = (__bf16*)alloc((size_t)M_ * D_ * sizeof(__bf16));
  __bf16* kh  = (__bf16*)alloc((size_t)M_ * D_ * sizeof(__bf16));
  __bf16* vtb = (__bf16*)alloc((size_t)M_ * D_ * sizeof(__bf16));
  __bf16* oh  = (__bf16*)alloc((size_t)M_ * D_ * sizeof(__bf16));

  // bf16 conversions
  int wn4 = E_ * D_ * D_ / 4;
  cvt_kernel<<<(wn4 + 255) / 256, 256, 0, stream>>>(wq, whq, wn4);
  cvt_kernel<<<(wn4 + 255) / 256, 256, 0, stream>>>(wk, whk, wn4);
  cvt_kernel<<<(wn4 + 255) / 256, 256, 0, stream>>>(wv, whv, wn4);
  cvt_kernel<<<(wn4 + 255) / 256, 256, 0, stream>>>(wo, who, wn4);
  int xn4 = M_ * D_ / 4;
  cvt_kernel<<<(xn4 + 255) / 256, 256, 0, stream>>>(x, xh, xn4);

  gate_kernel<<<M_, 128, 0, stream>>>(x, gate_w, gate_b, gtp);

  dim3 gg(D_ / 128, M_ / 128);
  for (int e = 0; e < E_; ++e) {
    const __bf16* wq_e = whq + (size_t)e * D_ * D_;
    const __bf16* wk_e = whk + (size_t)e * D_ * D_;
    const __bf16* wv_e = whv + (size_t)e * D_ * D_;
    const __bf16* wo_e = who + (size_t)e * D_ * D_;

    gemm_f32_kernel<<<gg, 256, 0, stream>>>(xh, wq_e, qf, D_, D_);
    gemm_f32_kernel<<<gg, 256, 0, stream>>>(xh, wk_e, kf, D_, D_);
    lnrope_kernel<<<M_, 256, 0, stream>>>(qf, q_gamma + e * D_, q_beta + e * D_,
                                          fcos, fsin, qh, 0.125f);   // 1/sqrt(64) folded into q
    lnrope_kernel<<<M_, 256, 0, stream>>>(kf, k_gamma + e * D_, k_beta + e * D_,
                                          fcos, fsin, kh, 1.0f);
    gemm_vt_kernel<<<gg, 256, 0, stream>>>(xh, wv_e, vtb, D_);
    flash_kernel<<<dim3(S_ / 64, H_, B_), 128, 0, stream>>>(qh, kh, vtb, oh);
    gemm_gated_kernel<<<gg, 256, 0, stream>>>(oh, wo_e, gtp, out, e, D_, D_);
  }
}
